// LinearAttention_53747220742256
// MI455X (gfx1250) — matrix operand order repaired and checked by reading, not compiled, not run on hardware
//
#include <hip/hip_runtime.h>

// ---------------- types ----------------
typedef __attribute__((ext_vector_type(16))) __bf16        v16bf;
typedef __attribute__((ext_vector_type(8)))  float         v8f;
typedef __attribute__((ext_vector_type(4)))  unsigned int  u32x4;

union Frag16 { v16bf v; u32x4 q[2]; };
union AccF   { v8f  v; float f[8]; };

__device__ __forceinline__ unsigned short to_bf16(float f) {
  unsigned int u = __float_as_uint(f);
  u += 0x7FFFu + ((u >> 16) & 1u);          // round-to-nearest-even
  return (unsigned short)(u >> 16);
}
__device__ __forceinline__ float from_bf16(unsigned int h) {
  return __uint_as_float((h & 0xFFFFu) << 16);
}

// A fragment: 16x32 bf16, row-major source with row stride lda (elements).
// Lane l (0..15): row=row0+l, K = k0+[0..15]
// Lane l (16..31): row=row0+(l-16), K = k0+[16..31]
__device__ __forceinline__ Frag16 load_a(const unsigned short* __restrict__ A,
                                         int lda, int row0, int k0, int lane) {
  Frag16 f;
  const size_t base = (size_t)(row0 + (lane & 15)) * lda + (k0 + ((lane >> 4) << 4));
  f.q[0] = *(const u32x4*)(A + base);
  f.q[1] = *(const u32x4*)(A + base + 8);
  return f;
}

// B fragment: 32x16 bf16 from transposed matrix Bt[out][in] (row stride ldb):
// element B[k][c] = Bt[c][k]. Lane l: col = col0+(l&15), K = k0+(l>>4)*16 + [0..15]
__device__ __forceinline__ Frag16 load_b(const unsigned short* __restrict__ Bt,
                                         int ldb, int col0, int k0, int lane) {
  Frag16 f;
  const size_t base = (size_t)(col0 + (lane & 15)) * ldb + (k0 + ((lane >> 4) << 4));
  const u32x4* p = (const u32x4*)(Bt + base);
  f.q[0] = p[0];
  f.q[1] = p[1];
  return f;
}

// LDS-staged B fragment. LDS slab: 64 cols x 64 K, row stride 72 (bank-conflict pad).
#define LDS_STRIDE 72
__device__ __forceinline__ Frag16 load_b_lds(const unsigned short* lds,
                                             int col0, int koff, int lane) {
  Frag16 f;
  const int c  = col0 + (lane & 15);
  const int kb = koff + ((lane >> 4) << 4);
  const u32x4* p = (const u32x4*)(lds + c * LDS_STRIDE + kb);
  f.q[0] = p[0];
  f.q[1] = p[1];
  return f;
}

// Stage 64 cols x 64 K of Bt (row stride ldb) starting at (nt, kk) into LDS.
// 256-thread variant: each thread moves 16 bf16 (32 B).
__device__ __forceinline__ void stage_b_256(unsigned short* ldsB,
                                            const unsigned short* __restrict__ Bt,
                                            int ldb, int nt, int kk, int tid) {
  const int c  = tid >> 2;            // 0..63
  const int ko = (tid & 3) << 4;      // 0,16,32,48
  const u32x4* g = (const u32x4*)(Bt + (size_t)(nt + c) * ldb + kk + ko);
  u32x4* s = (u32x4*)(ldsB + c * LDS_STRIDE + ko);
  s[0] = g[0];
  s[1] = g[1];
}

// 128-thread variant: each thread moves 32 bf16 (64 B).
__device__ __forceinline__ void stage_b_128(unsigned short* ldsB,
                                            const unsigned short* __restrict__ Bt,
                                            int ldb, int nt, int kk, int tid) {
  const int c  = tid >> 1;            // 0..63
  const int ko = (tid & 1) << 5;      // 0,32
  const u32x4* g = (const u32x4*)(Bt + (size_t)(nt + c) * ldb + kk + ko);
  u32x4* s = (u32x4*)(ldsB + c * LDS_STRIDE + ko);
  s[0] = g[0]; s[1] = g[1]; s[2] = g[2]; s[3] = g[3];
}

__device__ __forceinline__ v8f wmma_bf16(const Frag16& a, const Frag16& b, v8f c) {
  return __builtin_amdgcn_wmma_f32_16x16x32_bf16(false, a.v, false, b.v,
                                                 (short)0, c, false, false);
}

__device__ __forceinline__ float phi_elu1(float x) {   // elu(x)+1
  return x > 0.0f ? x + 1.0f : __expf(x);
}

// ---------------- constants ----------------
#define BATCH 8
#define NSEQ  4096
#define DIM   512
#define H3    1536
#define NHEAD 8
#define HDIM  64
#define MROWS (BATCH * NSEQ)          // 32768
#define BH    (BATCH * NHEAD)         // 64

// ---------------- convert kernels ----------------
__global__ __launch_bounds__(256) void cvt_x_kernel(const float* __restrict__ x,
                                                    unsigned short* __restrict__ xb) {
  size_t i = ((size_t)blockIdx.x * 256 + threadIdx.x) * 8;
  #pragma unroll
  for (int j = 0; j < 8; ++j) xb[i + j] = to_bf16(x[i + j]);
}

__global__ __launch_bounds__(256) void cvt_wqkv_kernel(const float* __restrict__ W,
                                                       unsigned short* __restrict__ Wt) {
  int idx = blockIdx.x * 256 + threadIdx.x;       // [H3*DIM)
  int o = idx >> 9, i = idx & 511;
  Wt[idx] = to_bf16(W[(size_t)i * H3 + o]);
}

__global__ __launch_bounds__(256) void cvt_wout_kernel(const float* __restrict__ W,
                                                       unsigned short* __restrict__ Wt) {
  int idx = blockIdx.x * 256 + threadIdx.x;       // [DIM*DIM)
  int o = idx >> 9, i = idx & 511;
  Wt[idx] = to_bf16(W[(size_t)i * DIM + o]);
}

// ---------------- fused qkv GEMM + feature map + scatter ----------------
// Block = 8 waves = 256 rows x 64 cols; each wave a 32x64 tile (8 accumulators).
// B slab (64 cols x 64 K) staged in LDS, shared by all 8 waves.
__global__ __launch_bounds__(256) void qkv_phi_kernel(
    const unsigned short* __restrict__ xb,     // [MROWS][DIM]
    const unsigned short* __restrict__ Wt,     // [H3][DIM] (transposed)
    unsigned short* __restrict__ q_phi,        // [BH][NSEQ][HDIM]
    unsigned short* __restrict__ k_phiT,       // [BH][HDIM][NSEQ]
    unsigned short* __restrict__ v_t)          // [BH][HDIM][NSEQ]
{
  __shared__ unsigned short ldsB[64 * LDS_STRIDE];
  const int tid  = threadIdx.x;
  const int lane = tid & 31;
  const int ww   = tid >> 5;                       // wave in block
  const int nstripe = blockIdx.x % 24;             // 24 column strips of 64
  const int mb      = blockIdx.x / 24;             // 128 row blocks of 256
  const int nt = nstripe * 64;
  const int mt = mb * 256 + ww * 32;

  AccF acc[2][4];
  #pragma unroll
  for (int s = 0; s < 2; ++s)
    #pragma unroll
    for (int t = 0; t < 4; ++t)
      #pragma unroll
      for (int j = 0; j < 8; ++j) acc[s][t].f[j] = 0.0f;

  for (int kk = 0; kk < DIM; kk += 64) {
    __syncthreads();
    stage_b_256(ldsB, Wt, DIM, nt, kk, tid);
    __syncthreads();
    if (kk + 64 < DIM)
      __builtin_prefetch(xb + (size_t)(mt + (lane & 15)) * DIM + kk + 64, 0, 1);
    #pragma unroll
    for (int k2 = 0; k2 < 64; k2 += 32) {
      Frag16 a0 = load_a(xb, DIM, mt,      kk + k2, lane);
      Frag16 a1 = load_a(xb, DIM, mt + 16, kk + k2, lane);
      #pragma unroll
      for (int t = 0; t < 4; ++t) {
        Frag16 b = load_b_lds(ldsB, t * 16, k2, lane);
        acc[0][t].v = wmma_bf16(a0, b, acc[0][t].v);
        acc[1][t].v = wmma_bf16(a1, b, acc[1][t].v);
      }
    }
  }

  const int seg    = nt >> 9;                  // 0=q, 1=k, 2=v
  const int colh   = nt & 511;                 // column within its 512 segment
  const int row_hi = (lane >> 4) << 3;
  const int ncol   = lane & 15;

  #pragma unroll
  for (int s = 0; s < 2; ++s) {
    #pragma unroll
    for (int t = 0; t < 4; ++t) {
      const int col = colh + t * 16 + ncol;    // strip is 64-aligned -> single head
      const int h = col >> 6, d = col & 63;
      #pragma unroll
      for (int j = 0; j < 8; ++j) {
        const int m = mt + s * 16 + row_hi + j;
        const int b = m >> 12, n = m & 4095;
        const int bh = b * NHEAD + h;
        const float fv = acc[s][t].f[j];
        if (seg == 0) {
          q_phi[((size_t)bh * NSEQ + n) * HDIM + d] = to_bf16(phi_elu1(fv));
        } else if (seg == 1) {
          k_phiT[((size_t)bh * HDIM + d) * NSEQ + n] = to_bf16(phi_elu1(fv));
        } else {
          v_t[((size_t)bh * HDIM + d) * NSEQ + n] = to_bf16(fv);
        }
      }
    }
  }
}

// ---------------- per-head KV = k_phi^T @ v  (M=64,N=64,K=4096) ----------------
// One block (4 waves) per (b,h); wave ww covers d rows [16*ww,16*ww+16).
// v slab (64 e-cols x 64 n) staged in LDS, shared by the 4 waves.
__global__ __launch_bounds__(128) void kv_kernel(
    const unsigned short* __restrict__ k_phiT,  // [BH][HDIM][NSEQ]
    const unsigned short* __restrict__ v_t,     // [BH][HDIM][NSEQ]
    unsigned short* __restrict__ KVb)           // [BH][e*64+d]
{
  __shared__ unsigned short ldsB[64 * LDS_STRIDE];
  const int tid  = threadIdx.x;
  const int lane = tid & 31;
  const int bh   = blockIdx.x;
  const int dt   = (tid >> 5) * 16;
  const unsigned short* A  = k_phiT + (size_t)bh * HDIM * NSEQ;
  const unsigned short* Bv = v_t    + (size_t)bh * HDIM * NSEQ;

  AccF acc[4];
  #pragma unroll
  for (int t = 0; t < 4; ++t)
    #pragma unroll
    for (int j = 0; j < 8; ++j) acc[t].f[j] = 0.0f;

  for (int kk = 0; kk < NSEQ; kk += 64) {
    __syncthreads();
    stage_b_128(ldsB, Bv, NSEQ, 0, kk, tid);
    __syncthreads();
    #pragma unroll
    for (int k2 = 0; k2 < 64; k2 += 32) {
      Frag16 a = load_a(A, NSEQ, dt, kk + k2, lane);
      #pragma unroll
      for (int t = 0; t < 4; ++t) {
        Frag16 b = load_b_lds(ldsB, t * 16, k2, lane);
        acc[t].v = wmma_bf16(a, b, acc[t].v);
      }
    }
  }

  const int row_hi = (lane >> 4) << 3;
  const int ncol   = lane & 15;
  #pragma unroll
  for (int t = 0; t < 4; ++t) {
    const int e = t * 16 + ncol;
    #pragma unroll
    for (int j = 0; j < 8; ++j) {
      const int d = dt + row_hi + j;
      KVb[(size_t)bh * 4096 + e * 64 + d] = to_bf16(acc[t].f[j]);  // stored [e][d]
    }
  }
}

// ---------------- K_sum[bh][d] = sum_n k_phi ----------------
__global__ __launch_bounds__(256) void ksum_kernel(const unsigned short* __restrict__ k_phiT,
                                                   float* __restrict__ K_sum) {
  const int idx = blockIdx.x * 256 + threadIdx.x;       // [BH*HDIM)
  const unsigned short* row = k_phiT + (size_t)idx * NSEQ;
  float s = 0.0f;
  for (int n = 0; n < NSEQ; n += 8) {
    u32x4 q = *(const u32x4*)(row + n);
    s += from_bf16(q.x) + from_bf16(q.x >> 16)
       + from_bf16(q.y) + from_bf16(q.y >> 16)
       + from_bf16(q.z) + from_bf16(q.z >> 16)
       + from_bf16(q.w) + from_bf16(q.w >> 16);
  }
  K_sum[idx] = s;
}

// ---------------- rdenom[bh][n] = 1 / (q_phi . K_sum + eps) ----------------
__global__ __launch_bounds__(256) void denom_kernel(const unsigned short* __restrict__ q_phi,
                                                    const float* __restrict__ K_sum,
                                                    float* __restrict__ rdenom) {
  const int idx = blockIdx.x * 256 + threadIdx.x;       // [BH*NSEQ)
  const int bh = idx >> 12, n = idx & 4095;
  const unsigned short* qrow = q_phi + ((size_t)bh * NSEQ + n) * HDIM;
  const float* ks = K_sum + bh * HDIM;
  float s = 0.0f;
  for (int d = 0; d < HDIM; d += 8) {
    u32x4 q = *(const u32x4*)(qrow + d);
    s += from_bf16(q.x)       * ks[d + 0];
    s += from_bf16(q.x >> 16) * ks[d + 1];
    s += from_bf16(q.y)       * ks[d + 2];
    s += from_bf16(q.y >> 16) * ks[d + 3];
    s += from_bf16(q.z)       * ks[d + 4];
    s += from_bf16(q.z >> 16) * ks[d + 5];
    s += from_bf16(q.w)       * ks[d + 6];
    s += from_bf16(q.w >> 16) * ks[d + 7];
  }
  rdenom[idx] = 1.0f / (s + 1e-6f);
}

// ---------------- attn[n][h*64+e] = (q_phi @ KV)[n][e] * rdenom[n] ----------------
__global__ __launch_bounds__(256) void attn_kernel(
    const unsigned short* __restrict__ q_phi,   // [BH][NSEQ][HDIM]
    const unsigned short* __restrict__ KVb,     // [BH][e*64+d]
    const float* __restrict__ rdenom,           // [BH][NSEQ]
    unsigned short* __restrict__ attn)          // [MROWS][DIM]
{
  const int lane = threadIdx.x & 31;
  const int w    = blockIdx.x * 8 + (threadIdx.x >> 5);
  const int bh   = w >> 8;
  const int mt   = (w & 255) * 16;
  const unsigned short* A  = q_phi + (size_t)bh * NSEQ * HDIM;
  const unsigned short* Bk = KVb   + (size_t)bh * 4096;

  AccF acc[4];
  #pragma unroll
  for (int t = 0; t < 4; ++t)
    #pragma unroll
    for (int j = 0; j < 8; ++j) acc[t].f[j] = 0.0f;

  #pragma unroll
  for (int kk = 0; kk < HDIM; kk += 32) {
    Frag16 a = load_a(A, HDIM, mt, kk, lane);
    #pragma unroll
    for (int t = 0; t < 4; ++t) {
      Frag16 b = load_b(Bk, HDIM, t * 16, kk, lane);   // B[d][e] = KVb[e*64+d]
      acc[t].v = wmma_bf16(a, b, acc[t].v);
    }
  }

  const int b_ = bh >> 3, h = bh & 7;
  const int row_hi = (lane >> 4) << 3;
  const int ncol   = lane & 15;
  #pragma unroll
  for (int j = 0; j < 8; ++j) {
    const int n = mt + row_hi + j;
    const float r = rdenom[bh * NSEQ + n];
    const size_t obase = ((size_t)(b_ * NSEQ + n)) * DIM + h * HDIM;
    #pragma unroll
    for (int t = 0; t < 4; ++t) {
      const int e = t * 16 + ncol;
      attn[obase + e] = to_bf16(acc[t].f[j] * r);
    }
  }
}

// ---------------- out = attn @ W_out + b_out (f32 output) ----------------
// Same blocking as qkv: 256 rows x 64 cols per block, 32x64 per wave, LDS-staged B.
__global__ __launch_bounds__(256) void out_kernel(
    const unsigned short* __restrict__ attn,    // [MROWS][DIM]
    const unsigned short* __restrict__ WoT,     // [DIM][DIM] transposed
    const float* __restrict__ bias,             // [DIM]
    float* __restrict__ out)                    // [MROWS][DIM]
{
  __shared__ unsigned short ldsB[64 * LDS_STRIDE];
  const int tid  = threadIdx.x;
  const int lane = tid & 31;
  const int ww   = tid >> 5;
  const int nstripe = blockIdx.x & 7;              // 8 column strips of 64
  const int mb      = blockIdx.x >> 3;             // 128 row blocks of 256
  const int nt = nstripe * 64;
  const int mt = mb * 256 + ww * 32;

  AccF acc[2][4];
  #pragma unroll
  for (int s = 0; s < 2; ++s)
    #pragma unroll
    for (int t = 0; t < 4; ++t)
      #pragma unroll
      for (int j = 0; j < 8; ++j) acc[s][t].f[j] = 0.0f;

  for (int kk = 0; kk < DIM; kk += 64) {
    __syncthreads();
    stage_b_256(ldsB, WoT, DIM, nt, kk, tid);
    __syncthreads();
    if (kk + 64 < DIM)
      __builtin_prefetch(attn + (size_t)(mt + (lane & 15)) * DIM + kk + 64, 0, 1);
    #pragma unroll
    for (int k2 = 0; k2 < 64; k2 += 32) {
      Frag16 a0 = load_a(attn, DIM, mt,      kk + k2, lane);
      Frag16 a1 = load_a(attn, DIM, mt + 16, kk + k2, lane);
      #pragma unroll
      for (int t = 0; t < 4; ++t) {
        Frag16 b = load_b_lds(ldsB, t * 16, k2, lane);
        acc[0][t].v = wmma_bf16(a0, b, acc[0][t].v);
        acc[1][t].v = wmma_bf16(a1, b, acc[1][t].v);
      }
    }
  }

  const int row_hi = (lane >> 4) << 3;
  const int ncol   = lane & 15;
  #pragma unroll
  for (int s = 0; s < 2; ++s) {
    #pragma unroll
    for (int t = 0; t < 4; ++t) {
      const int col = nt + t * 16 + ncol;
      const float bb = bias[col];
      #pragma unroll
      for (int j = 0; j < 8; ++j) {
        const int m = mt + s * 16 + row_hi + j;
        out[(size_t)m * DIM + col] = acc[s][t].f[j] + bb;
      }
    }
  }
}

// ---------------- launch ----------------
extern "C" void kernel_launch(void* const* d_in, const int* in_sizes, int n_in,
                              void* d_out, int out_size, void* d_ws, size_t ws_size,
                              hipStream_t stream) {
  const float* x     = (const float*)d_in[0];   // [8,4096,512]
  const float* W_qkv = (const float*)d_in[1];   // [512,1536]
  const float* W_out = (const float*)d_in[2];   // [512,512]
  const float* b_out = (const float*)d_in[3];   // [512]
  float* out = (float*)d_out;

  char* ws = (char*)d_ws;
  const size_t SZ_XB   = (size_t)MROWS * DIM * 2;   // 32 MiB (reused for attn)
  const size_t SZ_WQ   = (size_t)H3 * DIM * 2;
  const size_t SZ_WO   = (size_t)DIM * DIM * 2;
  const size_t SZ_PHI  = (size_t)BH * NSEQ * HDIM * 2;
  const size_t SZ_KVB  = (size_t)BH * 4096 * 2;
  const size_t SZ_KSUM = (size_t)BH * HDIM * 4;

  unsigned short* xb     = (unsigned short*)(ws);
  unsigned short* wqkv_t = (unsigned short*)(ws + SZ_XB);
  unsigned short* wout_t = (unsigned short*)(ws + SZ_XB + SZ_WQ);
  unsigned short* q_phi  = (unsigned short*)(ws + SZ_XB + SZ_WQ + SZ_WO);
  unsigned short* k_phiT = (unsigned short*)(ws + SZ_XB + SZ_WQ + SZ_WO + SZ_PHI);
  unsigned short* v_t    = (unsigned short*)(ws + SZ_XB + SZ_WQ + SZ_WO + 2 * SZ_PHI);
  unsigned short* kvb    = (unsigned short*)(ws + SZ_XB + SZ_WQ + SZ_WO + 3 * SZ_PHI);
  float*          ksum   = (float*)(ws + SZ_XB + SZ_WQ + SZ_WO + 3 * SZ_PHI + SZ_KVB);
  float*          rden   = (float*)(ws + SZ_XB + SZ_WQ + SZ_WO + 3 * SZ_PHI + SZ_KVB + SZ_KSUM);
  unsigned short* attn   = xb;  // xb dead after qkv_phi_kernel; reuse for attn

  cvt_x_kernel   <<<8192, 256, 0, stream>>>(x, xb);
  cvt_wqkv_kernel<<<3072, 256, 0, stream>>>(W_qkv, wqkv_t);
  cvt_wout_kernel<<<1024, 256, 0, stream>>>(W_out, wout_t);

  qkv_phi_kernel <<<3072, 256, 0, stream>>>(xb, wqkv_t, q_phi, k_phiT, v_t);

  kv_kernel      <<<BH,   128, 0, stream>>>(k_phiT, v_t, kvb);
  ksum_kernel    <<<16,   256, 0, stream>>>(k_phiT, ksum);
  denom_kernel   <<<1024, 256, 0, stream>>>(q_phi, ksum, rden);

  attn_kernel    <<<2048, 256, 0, stream>>>(q_phi, kvb, rden, attn);

  out_kernel     <<<1024, 256, 0, stream>>>(attn, wout_t, b_out, out);
}
